// DataTransformer_2757369004744
// MI455X (gfx1250) — compile-verified, matrix operand order checked by reading, not run on hardware
//
#include <hip/hip_runtime.h>

// MI455X (gfx1250) fused gated multi-transform kernel.
//
// out[r,:] = sum_t softmax_t( (x_r . p_t) / (|x_r||p_t|) ) * (x_r @ W_t^T + b_t) + x_r
//
// Strategy (see analysis): fully fused, bf16 WMMA with hi/lo split-precision
// (3 products) for ~f32 accuracy; W converted once to bf16 hi/lo in d_ws;
// A tile staged in LDS; per-row softmax gate applied in registers.
//
// ws layout: [ Whi: T*D*D bf16 | Wlo: T*D*D bf16 | inv_pnorm: T f32 ]  (~512KB)

typedef __attribute__((ext_vector_type(16))) __bf16 v16bf;
typedef __attribute__((ext_vector_type(8)))  __bf16 v8bf;
typedef __attribute__((ext_vector_type(8)))  float  v8f;

#define B_  32
#define S_  2048
#define D_  128
#define T_  8
#define NROWS (B_ * S_)
#define ROWS_PER_WG 32

__device__ __forceinline__ v8f wmma_bf16(v16bf a, v16bf b, v8f c) {
  // (neg_a, A, neg_b, B, c_mod, C, reuse_a, reuse_b)
  return __builtin_amdgcn_wmma_f32_16x16x32_bf16(false, a, false, b, (short)0, c,
                                                 false, false);
}

// Build a 16x32 (or Kx16) bf16 fragment for this lane from two contiguous
// 8-element (16B) chunks at k and k+16 (ISA 16-bit matrix lane layout).
__device__ __forceinline__ v16bf load_frag(const __bf16* base) {
  v8bf lo = *(const v8bf*)base;         // K = kb+koff   .. +7
  v8bf hi = *(const v8bf*)(base + 16);  // K = kb+koff+16.. +23
  return __builtin_shufflevector(lo, hi, 0, 1, 2, 3, 4, 5, 6, 7,
                                         8, 9, 10, 11, 12, 13, 14, 15);
}

// ---------------------------------------------------------------------------
// Prep: W f32 -> bf16 hi/lo split; prototype inverse norms.
// ---------------------------------------------------------------------------
__global__ __launch_bounds__(256) void dt_prep_weights(
    const float* __restrict__ Wx, const float* __restrict__ p,
    __bf16* __restrict__ whi, __bf16* __restrict__ wlo,
    float* __restrict__ inv_pnorm) {
  int i = blockIdx.x * 256 + threadIdx.x;
  if (i < T_ * D_ * D_) {
    float w   = Wx[i];
    __bf16 h  = (__bf16)w;
    whi[i]    = h;
    wlo[i]    = (__bf16)(w - (float)h);
  }
  if (blockIdx.x == 0 && threadIdx.x < T_) {
    const float* pr = p + threadIdx.x * D_;
    float s = 0.f;
    for (int d = 0; d < D_; ++d) s += pr[d] * pr[d];
    inv_pnorm[threadIdx.x] = rsqrtf(s);
  }
}

// ---------------------------------------------------------------------------
// Fused gate + 8-way GEMM + combine + residual.
// 256 threads = 8 wave32 waves; WG tile = 32 rows x 128 cols.
// Wave w owns output columns [16w, 16w+16); handles 2 m-tiles of 16 rows.
// ---------------------------------------------------------------------------
__global__ __launch_bounds__(256) void dt_fused(
    const float* __restrict__ x, const float* __restrict__ bx,
    const float* __restrict__ p,
    const __bf16* __restrict__ whi, const __bf16* __restrict__ wlo,
    const float* __restrict__ inv_pnorm, float* __restrict__ out) {
  __shared__ __align__(16) float  Af [ROWS_PER_WG][132];  // padded: bank spread
  __shared__ __align__(16) __bf16 Ahi[ROWS_PER_WG][136];
  __shared__ __align__(16) __bf16 Alo[ROWS_PER_WG][136];
  __shared__ float Cos[ROWS_PER_WG][T_];
  __shared__ float Sim[ROWS_PER_WG][T_];

  const int  tid     = threadIdx.x;
  const long rowBase = (long)blockIdx.x * ROWS_PER_WG;

  // ---- stage A tile: f32 copy + bf16 hi/lo split (coalesced 64B/thread) ----
  {
    const int r  = tid >> 3;
    const int c0 = (tid & 7) * 16;
    const float* src = x + (rowBase + r) * D_ + c0;
#pragma unroll
    for (int j = 0; j < 16; j += 4) {
      float4 v = *(const float4*)(src + j);
      float vv[4] = {v.x, v.y, v.z, v.w};
#pragma unroll
      for (int q = 0; q < 4; ++q) {
        float w = vv[q];
        Af[r][c0 + j + q] = w;
        __bf16 h = (__bf16)w;
        Ahi[r][c0 + j + q] = h;
        Alo[r][c0 + j + q] = (__bf16)(w - (float)h);
      }
    }
  }
  __syncthreads();

  // ---- gating: one (row, t) pair per thread ----
  {
    const int r = tid >> 3, t = tid & 7;
    float dot = 0.f, ss = 0.f;
    for (int c = 0; c < D_; c += 4) {
      float4 a  = *(const float4*)&Af[r][c];
      float4 pv = *(const float4*)(p + t * D_ + c);
      dot += a.x * pv.x + a.y * pv.y + a.z * pv.z + a.w * pv.w;
      ss  += a.x * a.x + a.y * a.y + a.z * a.z + a.w * a.w;
    }
    Cos[r][t] = dot * rsqrtf(ss) * inv_pnorm[t];
  }
  __syncthreads();
  {
    const int r = tid >> 3, t = tid & 7;
    float m = Cos[r][0];
#pragma unroll
    for (int i = 1; i < T_; ++i) m = fmaxf(m, Cos[r][i]);
    float sum = 0.f;
#pragma unroll
    for (int i = 0; i < T_; ++i) sum += __expf(Cos[r][i] - m);
    Sim[r][t] = __expf(Cos[r][t] - m) / sum;
  }
  __syncthreads();

  // ---- WMMA GEMMs ----
  const int wave = tid >> 5;      // n-tile id (0..7)
  const int lane = tid & 31;
  const int lrow = lane & 15;     // row/col within 16-tile
  const int hi16 = lane >> 4;     // lane half
  const int koff = hi16 * 8;      // per-half K offset (ISA 16-bit layout)
  const int ncol = wave * 16 + lrow;

  const v8f vzero = {0.f, 0.f, 0.f, 0.f, 0.f, 0.f, 0.f, 0.f};
  v8f outAcc0 = vzero, outAcc1 = vzero;

  for (int t = 0; t < T_; ++t) {
    v8f acc0 = vzero, acc1 = vzero;
    for (int ks = 0; ks < 4; ++ks) {
      const int kb = ks * 32;
      // B = W_t^T : B[k][n] = W_t[n][k]; lane holds column n = ncol.
      const __bf16* bh = whi + (((size_t)t * D_ + ncol) * D_ + kb + koff);
      const __bf16* bl = wlo + (((size_t)t * D_ + ncol) * D_ + kb + koff);
      v16bf Bh = load_frag(bh);
      v16bf Bl = load_frag(bl);

      // m-tile 0 (rows 0..15)
      {
        v16bf Ah = load_frag(&Ahi[lrow][kb + koff]);
        v16bf Al = load_frag(&Alo[lrow][kb + koff]);
        acc0 = wmma_bf16(Ah, Bh, acc0);
        acc0 = wmma_bf16(Ah, Bl, acc0);
        acc0 = wmma_bf16(Al, Bh, acc0);
      }
      // m-tile 1 (rows 16..31)
      {
        v16bf Ah = load_frag(&Ahi[16 + lrow][kb + koff]);
        v16bf Al = load_frag(&Alo[16 + lrow][kb + koff]);
        acc1 = wmma_bf16(Ah, Bh, acc1);
        acc1 = wmma_bf16(Ah, Bl, acc1);
        acc1 = wmma_bf16(Al, Bh, acc1);
      }
    }
    // Weighted combine: out += sim[row][t] * (acc + bias_t[ncol]).
    // C/D layout: VGPR v, lanes 0-15 -> M=v ; lanes 16-31 -> M=v+8.
    const float bias = bx[t * D_ + ncol];
    const int   rb   = hi16 * 8;
#pragma unroll
    for (int v = 0; v < 8; ++v) {
      outAcc0[v] += Sim[rb + v][t]      * (acc0[v] + bias);
      outAcc1[v] += Sim[16 + rb + v][t] * (acc1[v] + bias);
    }
  }

  // ---- residual + store ----
#pragma unroll
  for (int v = 0; v < 8; ++v) {
    const int r0 = hi16 * 8 + v;
    const int r1 = 16 + r0;
    out[(rowBase + r0) * D_ + ncol] = outAcc0[v] + Af[r0][ncol];
    out[(rowBase + r1) * D_ + ncol] = outAcc1[v] + Af[r1][ncol];
  }
}

// ---------------------------------------------------------------------------
extern "C" void kernel_launch(void* const* d_in, const int* in_sizes, int n_in,
                              void* d_out, int out_size, void* d_ws,
                              size_t ws_size, hipStream_t stream) {
  (void)in_sizes; (void)n_in; (void)out_size; (void)ws_size;

  const float* x  = (const float*)d_in[0];  // [B,S,D] f32
  const float* Wx = (const float*)d_in[1];  // [T,D,D] f32
  const float* bx = (const float*)d_in[2];  // [T,D]   f32
  const float* p  = (const float*)d_in[3];  // [T,1,D] f32
  float*       o  = (float*)d_out;          // [B,S,D] f32

  __bf16* whi = (__bf16*)d_ws;
  __bf16* wlo = whi + (size_t)T_ * D_ * D_;
  float*  ipn = (float*)(wlo + (size_t)T_ * D_ * D_);

  dt_prep_weights<<<(T_ * D_ * D_ + 255) / 256, 256, 0, stream>>>(Wx, p, whi,
                                                                  wlo, ipn);
  dt_fused<<<NROWS / ROWS_PER_WG, 256, 0, stream>>>(x, bx, p, whi, wlo, ipn, o);
}
